// DCGRUCell_37306085933702
// MI455X (gfx1250) — matrix operand order repaired and checked by reading, not compiled, not run on hardware
//
#include <hip/hip_runtime.h>
#include <hip/hip_bf16.h>
#include <math.h>

// ---------------------------------------------------------------------------
// DCGRU cell for MI455X (gfx1250).
// Dominant cost: 4x dense A[10000x10000] @ X[10000x2112] GEMMs (1.69 TFLOP).
// Done with V_WMMA_F32_16X16X32_BF16 using a split-bf16 3-product scheme
// (Ahi@Xhi + Ahi@Xlo + Alo@Xhi, fp32 accumulate) -> ~fp32 accuracy at ~2.7x
// fewer matrix instructions than the f32 K=4 WMMA path. hi/lo split happens
// in-register during LDS staging; HBM traffic & workspace unchanged.
// Single barrier per K-chunk (2-deep LDS pipeline), wide b128 LDS stores.
// ---------------------------------------------------------------------------

typedef __attribute__((ext_vector_type(16))) __bf16 v16bf;
typedef __attribute__((ext_vector_type(8)))  __bf16 v8bf;
typedef __attribute__((ext_vector_type(8)))  float  v8f;

#define NN 10000          // graph nodes
#define BB 32             // batch
#define DIN 2             // input dim
#define HH 64             // hidden
#define GIN 66            // DIN + HH
#define CC 2112           // GIN * BB  (columns of diffusion GEMMs)
#define FF 198            // GIN * 3   (projection features)

#define BM 128            // GEMM block tile M; 10000 = 78*128 + 16 (guarded stores)
#define BN 64             // GEMM block tile N; 2112 = 33*64 exact
#define BK 32             // GEMM K chunk (one bf16 WMMA depth)
#define NKC 313           // ceil(10000/32); last chunk has 16 valid K (zero-padded)
#define PAB 40            // bf16 pitch for A tile rows (80 B -> 16B-aligned runs)
#define PXB 40            // bf16 pitch for X tile cols (transposed)

union FragBF { v16bf v; v8bf h[2]; };

// ---------------------------------------------------------------------------
// Build X0 [N, C] with c = d*B + b:  x0[n, d*32+b] = (d<2 ? inputs : (r?)*states)
// ---------------------------------------------------------------------------
__global__ __launch_bounds__(256) void pack_x0_kernel(
    const float* __restrict__ inputs, const float* __restrict__ states,
    const float* __restrict__ rgate, float* __restrict__ X0, int use_r)
{
  long idx = (long)blockIdx.x * blockDim.x + threadIdx.x;
  if (idx >= (long)NN * CC) return;
  int n   = (int)(idx / CC);
  int rem = (int)(idx - (long)n * CC);
  int d = rem >> 5;          // 0..65
  int b = rem & 31;
  float v;
  if (d < DIN) {
    v = inputs[((size_t)b * NN + n) * DIN + d];
  } else {
    size_t o = ((size_t)b * NN + n) * HH + (d - DIN);
    v = states[o];
    if (use_r) v *= rgate[o];
  }
  X0[idx] = v;
}

// ---------------------------------------------------------------------------
// Y[i,c] = alpha * sum_k A[i,k]*X[k,c]  (+ beta * Z[i,c] if useZ)
// Split-bf16 WMMA: block 256 thr = 8 waves (4 along M x 2 along N),
// wave tile 32x32 = 2x2 subtiles, 3 bf16 WMMAs per subtile per 32-K chunk.
// Frag layouts (16-bit, wave32):
//   A: lane = row L%16; K runs [8h..8h+7] and [16+8h..16+8h+7], h = L/16
//   B: lane = col L%16; K runs [8h..8h+7] and [16+8h..16+8h+7] (same as A)
//   D: vgpr v -> M = v + 8h, N = L%16
// ---------------------------------------------------------------------------
__global__ __launch_bounds__(256) void gemm_hop_kernel(
    const float* __restrict__ A, const float* __restrict__ X,
    const float* __restrict__ Z, float* __restrict__ Y,
    float alpha, float beta, int useZ)
{
  __shared__ __bf16 Ah[2][BM * PAB];
  __shared__ __bf16 Al[2][BM * PAB];
  __shared__ __bf16 Xh[2][BN * PXB];
  __shared__ __bf16 Xl[2][BN * PXB];

  const int tid  = threadIdx.x;
  const int lane = tid & 31;
  const int wave = tid >> 5;      // 0..7
  const int wm   = wave >> 1;     // 0..3 (M)
  const int wn   = wave & 1;      // 0..1 (N)
  const int row0 = blockIdx.y * BM;
  const int col0 = blockIdx.x * BN;

  const v8f vzero = {0.f,0.f,0.f,0.f,0.f,0.f,0.f,0.f};
  v8f acc[2][2];
  acc[0][0] = vzero; acc[0][1] = vzero; acc[1][0] = vzero; acc[1][1] = vzero;

  // global->register staging:
  //   A 128x32 f32: 16/thr, row-contiguous (b128 x4)
  //   X 32x64 f32:  8/thr, one column x 8 k (coalesced across 64 lanes per k-row)
  const int arow = tid >> 1;        // 0..127
  const int akq  = (tid & 1) * 16;  // 0 or 16
  const int xc   = tid & 63;        // 0..63 (column)
  const int xkq  = (tid >> 6) * 8;  // 0,8,16,24
  // clamp A row: padded rows compute garbage that is never stored
  const int ar = (row0 + arow < NN) ? (row0 + arow) : (NN - 1);

  float a_reg[16];
  float x_reg[8];

  auto g_load = [&](int k0, bool tail) {
    if (!tail) {
      const float* p = A + (size_t)ar * NN + (k0 + akq);
#pragma unroll
      for (int i = 0; i < 16; ++i) a_reg[i] = p[i];
      __builtin_prefetch((const void*)(p + BK), 0, 1);   // chunk-after-next of A stream
      const float* q = X + (size_t)(k0 + xkq) * CC + (col0 + xc);
#pragma unroll
      for (int i = 0; i < 8; ++i) x_reg[i] = q[(size_t)i * CC];
    } else {
      const bool av = (k0 + akq) < NN;           // whole 16-run uniformly valid or not
      const float* p = A + (size_t)ar * NN + (av ? (k0 + akq) : 0);
#pragma unroll
      for (int i = 0; i < 16; ++i) a_reg[i] = av ? p[i] : 0.f;
      const bool xv = (k0 + xkq) < NN;           // whole 8-run uniformly valid or not
      const float* q = X + (size_t)(xv ? (k0 + xkq) : 0) * CC + (col0 + xc);
#pragma unroll
      for (int i = 0; i < 8; ++i) x_reg[i] = xv ? q[(size_t)i * CC] : 0.f;
    }
  };

  auto lds_store = [&](int buf) {
    v8bf h0, h1, l0, l1;
#pragma unroll
    for (int i = 0; i < 8; ++i) {
      float v = a_reg[i];
      __bf16 hb = (__bf16)v;
      h0[i] = hb; l0[i] = (__bf16)(v - (float)hb);
    }
#pragma unroll
    for (int i = 0; i < 8; ++i) {
      float v = a_reg[8 + i];
      __bf16 hb = (__bf16)v;
      h1[i] = hb; l1[i] = (__bf16)(v - (float)hb);
    }
    *(v8bf*)&Ah[buf][arow * PAB + akq]     = h0;
    *(v8bf*)&Ah[buf][arow * PAB + akq + 8] = h1;
    *(v8bf*)&Al[buf][arow * PAB + akq]     = l0;
    *(v8bf*)&Al[buf][arow * PAB + akq + 8] = l1;
    // X transposed: Xs[c][k]; one aligned 16B store per plane
    v8bf xh_v, xl_v;
#pragma unroll
    for (int i = 0; i < 8; ++i) {
      float v = x_reg[i];
      __bf16 hb = (__bf16)v;
      xh_v[i] = hb; xl_v[i] = (__bf16)(v - (float)hb);
    }
    *(v8bf*)&Xh[buf][xc * PXB + xkq] = xh_v;
    *(v8bf*)&Xl[buf][xc * PXB + xkq] = xl_v;
  };

  const int frow = lane & 15;
  const int half = lane >> 4;       // 0/1

  g_load(0, false);

  // 2-deep pipeline, ONE barrier per chunk:
  //   store(buf) ; stage chunk kc+1 ; barrier ; compute(buf)
  // store(kc+1) writes buf^1 which compute(kc) never reads; every
  // cross-iteration RAW/WAR pair is separated by exactly one barrier.
  for (int kc = 0; kc < NKC; ++kc) {
    const int buf = kc & 1;
    lds_store(buf);
    if (kc + 1 < NKC) g_load((kc + 1) * BK, (kc + 1) == NKC - 1);
    __syncthreads();

    const __bf16* ah = Ah[buf];
    const __bf16* al = Al[buf];
    const __bf16* xh = Xh[buf];
    const __bf16* xl = Xl[buf];

    FragBF a_h[2], a_l[2], b_h[2], b_l[2];
#pragma unroll
    for (int mi = 0; mi < 2; ++mi) {
      const int rr = (wm * 32 + mi * 16 + frow) * PAB;
      a_h[mi].h[0] = *(const v8bf*)&ah[rr + 8 * half];
      a_h[mi].h[1] = *(const v8bf*)&ah[rr + 16 + 8 * half];
      a_l[mi].h[0] = *(const v8bf*)&al[rr + 8 * half];
      a_l[mi].h[1] = *(const v8bf*)&al[rr + 16 + 8 * half];
    }
#pragma unroll
    for (int ni = 0; ni < 2; ++ni) {
      const int cr = (wn * 32 + ni * 16 + frow) * PXB;
      b_h[ni].h[0] = *(const v8bf*)&xh[cr + 8 * half];
      b_h[ni].h[1] = *(const v8bf*)&xh[cr + 16 + 8 * half];
      b_l[ni].h[0] = *(const v8bf*)&xl[cr + 8 * half];
      b_l[ni].h[1] = *(const v8bf*)&xl[cr + 16 + 8 * half];
    }
#pragma unroll
    for (int mi = 0; mi < 2; ++mi) {
#pragma unroll
      for (int ni = 0; ni < 2; ++ni) {
        acc[mi][ni] = __builtin_amdgcn_wmma_f32_16x16x32_bf16(
            false, a_h[mi].v, false, b_h[ni].v, (short)0, acc[mi][ni], false, false);
        acc[mi][ni] = __builtin_amdgcn_wmma_f32_16x16x32_bf16(
            false, a_h[mi].v, false, b_l[ni].v, (short)0, acc[mi][ni], false, false);
        acc[mi][ni] = __builtin_amdgcn_wmma_f32_16x16x32_bf16(
            false, a_l[mi].v, false, b_h[ni].v, (short)0, acc[mi][ni], false, false);
      }
    }
  }

  // Epilogue: D frag -> vgpr v: M = v + 8*(lane/16), N = lane%16
  const int moff = half * 8;
#pragma unroll
  for (int mi = 0; mi < 2; ++mi) {
#pragma unroll
    for (int ni = 0; ni < 2; ++ni) {
      const int rbase = row0 + wm * 32 + mi * 16 + moff;
      const int c     = col0 + wn * 32 + ni * 16 + frow;
#pragma unroll
      for (int v = 0; v < 8; ++v) {
        const int r = rbase + v;
        if (r < NN) {
          const size_t off = (size_t)r * CC + c;
          float val = alpha * acc[mi][ni][v];
          if (useZ) val += beta * Z[off];
          Y[off] = val;
        }
      }
    }
  }
}

// ---------------------------------------------------------------------------
// ru = sigmoid(Xk @ W_ru + b_ru); r = ru[:,:64], u = ru[:,64:]
// One block per (b,n) row; feature f = d*3 + m gathered from xs0/xs1/xs2.
// ---------------------------------------------------------------------------
__global__ __launch_bounds__(128) void project_ru_kernel(
    const float* __restrict__ X0f, const float* __restrict__ X1f,
    const float* __restrict__ X2f, const float* __restrict__ W,
    const float* __restrict__ bias, float* __restrict__ r, float* __restrict__ u)
{
  __shared__ float feat[FF];
  const int row = blockIdx.x;           // b*N + n
  const int b = row / NN;
  const int n = row - b * NN;
  const int t = threadIdx.x;            // 0..127 output channel

  if (t < FF) {
    const int d = t / 3, m = t - d * 3;
    const float* Xm = (m == 0) ? X0f : (m == 1) ? X1f : X2f;
    feat[t] = Xm[(size_t)n * CC + d * BB + b];
  }
  __syncthreads();

  float acc = bias[t];
#pragma unroll 6
  for (int f = 0; f < FF; ++f) acc += feat[f] * W[f * 128 + t];
  const float s = 1.0f / (1.0f + __expf(-acc));

  const size_t o = (size_t)row * HH;
  if (t < HH) r[o + t] = s;
  else        u[o + (t - HH)] = s;
}

// ---------------------------------------------------------------------------
// c = tanh(Xk2 @ W_c + b_c); out = u*states + (1-u)*c
// ---------------------------------------------------------------------------
__global__ __launch_bounds__(64) void project_c_kernel(
    const float* __restrict__ X0f, const float* __restrict__ X1f,
    const float* __restrict__ X2f, const float* __restrict__ W,
    const float* __restrict__ bias, const float* __restrict__ u,
    const float* __restrict__ states, float* __restrict__ out)
{
  __shared__ float feat[FF];
  const int row = blockIdx.x;           // b*N + n
  const int b = row / NN;
  const int n = row - b * NN;
  const int t = threadIdx.x;            // 0..63 output channel

  for (int i = t; i < FF; i += 64) {
    const int d = i / 3, m = i - d * 3;
    const float* Xm = (m == 0) ? X0f : (m == 1) ? X1f : X2f;
    feat[i] = Xm[(size_t)n * CC + d * BB + b];
  }
  __syncthreads();

  float acc = bias[t];
#pragma unroll 6
  for (int f = 0; f < FF; ++f) acc += feat[f] * W[f * HH + t];
  const float c = tanhf(acc);

  const size_t o = (size_t)row * HH + t;   // states/out are [(b*N+n)*H + h]
  const float uu = u[o];
  out[o] = uu * states[o] + (1.0f - uu) * c;
}

// ---------------------------------------------------------------------------
extern "C" void kernel_launch(void* const* d_in, const int* in_sizes, int n_in,
                              void* d_out, int out_size, void* d_ws, size_t ws_size,
                              hipStream_t stream) {
  (void)in_sizes; (void)n_in; (void)out_size; (void)ws_size;
  const float* inputs = (const float*)d_in[0];
  const float* states = (const float*)d_in[1];
  const float* A      = (const float*)d_in[2];
  const float* W_ru   = (const float*)d_in[3];
  const float* b_ru   = (const float*)d_in[4];
  const float* W_c    = (const float*)d_in[5];
  const float* b_c    = (const float*)d_in[6];
  float* out = (float*)d_out;

  // workspace layout (floats): X0,X1,X2 [N*C] reused across both gconvs; r,u [B*N*H]
  float* X0 = (float*)d_ws;
  float* X1 = X0 + (size_t)NN * CC;
  float* X2 = X1 + (size_t)NN * CC;
  float* rb = X2 + (size_t)NN * CC;
  float* ub = rb + (size_t)BB * NN * HH;

  const dim3 gpack((unsigned)(((long)NN * CC + 255) / 256));
  const dim3 ggemm(CC / BN, (NN + BM - 1) / BM);   // (33, 79): col-tiles adjacent -> A panel stays in L2

  // ---- gconv 1: ru gates ----
  pack_x0_kernel<<<gpack, 256, 0, stream>>>(inputs, states, nullptr, X0, 0);
  gemm_hop_kernel<<<ggemm, 256, 0, stream>>>(A, X0, X0, X1, 1.0f,  0.0f, 0); // X1 = A@X0
  gemm_hop_kernel<<<ggemm, 256, 0, stream>>>(A, X1, X0, X2, 2.0f, -1.0f, 1); // X2 = 2A@X1 - X0
  project_ru_kernel<<<BB * NN, 128, 0, stream>>>(X0, X1, X2, W_ru, b_ru, rb, ub);

  // ---- gconv 2: candidate ----
  pack_x0_kernel<<<gpack, 256, 0, stream>>>(inputs, states, rb, X0, 1);
  gemm_hop_kernel<<<ggemm, 256, 0, stream>>>(A, X0, X0, X1, 1.0f,  0.0f, 0);
  gemm_hop_kernel<<<ggemm, 256, 0, stream>>>(A, X1, X0, X2, 2.0f, -1.0f, 1);
  project_c_kernel<<<BB * NN, 64, 0, stream>>>(X0, X1, X2, W_c, b_c, ub, states, out);
}
